// GraphConstructor_62740882260605
// MI455X (gfx1250) — matrix operand order repaired and checked by reading, not compiled, not run on hardware
//
#include <hip/hip_runtime.h>
#include <hip/hip_bf16.h>

// ---------------------------------------------------------------------------
// Types for CDNA5 WMMA (wave32): v16bf A/B fragments (8 VGPRs), v8f C/D.
// ---------------------------------------------------------------------------
typedef unsigned short u16;
typedef __attribute__((ext_vector_type(8)))  u16    v8us;
typedef __attribute__((ext_vector_type(16))) u16    v16us;
typedef __attribute__((ext_vector_type(16))) __bf16 v16bf;
typedef __attribute__((ext_vector_type(8)))  float  v8f;

static constexpr int N_NODES = 2048;
static constexpr int BATCH   = 8;
static constexpr int SEQ     = 12;
static constexpr int HID     = 64;   // H
static constexpr int C2H     = 128;  // 2H
static constexpr int ROWS    = BATCH * N_NODES; // 16384

__device__ __forceinline__ u16 f2bf(float f) {
  unsigned u = __builtin_bit_cast(unsigned, f);
  u += 0x7FFFu + ((u >> 16) & 1u);          // round-to-nearest-even
  return (u16)(u >> 16);
}

// 16-bit A-matrix 16x32 fragment (ISA 7.12.2): element i of lane L holds
// k = kofs + (i/8)*16 + (L>=16 ? 8 : 0) + (i%8), row M = L&15.
// => two contiguous 16-byte loads per lane.
__device__ __forceinline__ v16bf load_a_frag(const u16* rowbase, int lane, int kofs) {
  int kb = kofs + ((lane >> 4) & 1) * 8;
  v8us lo = *(const v8us*)(rowbase + kb);
  v8us hi = *(const v8us*)(rowbase + kb + 16);
  v16us c;
#pragma unroll
  for (int i = 0; i < 8; ++i) { c[i] = lo[i]; c[i + 8] = hi[i]; }
  return __builtin_bit_cast(v16bf, c);
}

// 16-bit B-matrix 32x16 fragment: element i of lane L holds
// k = kofs + (i/8)*16 + (L>=16 ? 8 : 0) + (i%8), column N = L&15 (B column == K-matrix row).
// => two contiguous 16-byte loads per lane.
__device__ __forceinline__ v16bf load_b_frag(const u16* colrow, int lane, int kofs) {
  int kb = kofs + ((lane >> 4) & 1) * 8;
  v8us lo = *(const v8us*)(colrow + kb);
  v8us hi = *(const v8us*)(colrow + kb + 16);
  v16us c;
#pragma unroll
  for (int i = 0; i < 8; ++i) { c[i] = lo[i]; c[i + 8] = hi[i]; }
  return __builtin_bit_cast(v16bf, c);
}

// ---------------------------------------------------------------------------
// 1) ns = tanh(3*(emb_s@lin1.T+b1)), nt = tanh(3*(emb_t@lin2.T+b2)); + bf16.
// ---------------------------------------------------------------------------
__global__ void embed_kernel(const float* __restrict__ emb_s, const float* __restrict__ emb_t,
                             const float* __restrict__ l1w, const float* __restrict__ l1b,
                             const float* __restrict__ l2w, const float* __restrict__ l2b,
                             float* __restrict__ nsf, float* __restrict__ ntf,
                             u16* __restrict__ nsbf, u16* __restrict__ ntbf) {
  int n = blockIdx.x;
  int t = threadIdx.x;              // 64 threads: 0..31 -> ns, 32..63 -> nt
  bool is_t = t >= 32;
  int d = t & 31;
  const float* emb = is_t ? emb_t : emb_s;
  const float* w   = is_t ? l2w   : l1w;
  const float* bia = is_t ? l2b   : l1b;
  const float* er = emb + n * 32;
  const float* wr = w + d * 32;
  float acc = bia[d];
#pragma unroll
  for (int k = 0; k < 32; ++k) acc += er[k] * wr[k];
  float v = tanhf(3.0f * acc);
  if (is_t) { ntf[n * 32 + d] = v; ntbf[n * 32 + d] = f2bf(v); }
  else      { nsf[n * 32 + d] = v; nsbf[n * 32 + d] = f2bf(v); }
}

// ---------------------------------------------------------------------------
// 2) adj = relu(tanh(3*(ns@nt.T - nt@ns.T))) via bf16 WMMA (K=32 = one WMMA).
//    8 waves/block, each wave owns one 16x16 tile; block covers 16x128.
// ---------------------------------------------------------------------------
__global__ void adj_gemm_kernel(const u16* __restrict__ nsbf, const u16* __restrict__ ntbf,
                                float* __restrict__ adj) {
  const int N = N_NODES;
  int lane = threadIdx.x & 31, w = threadIdx.x >> 5;
  int r0 = blockIdx.y * 16;
  int c0 = blockIdx.x * 128 + w * 16;
  const u16* nsrow = nsbf + (r0 + (lane & 15)) * 32;
  const u16* ntrow = ntbf + (r0 + (lane & 15)) * 32;
  v16bf aS = load_a_frag(nsrow, lane, 0);
  v16bf aT = load_a_frag(ntrow, lane, 0);
  const u16* nscol = nsbf + (c0 + (lane & 15)) * 32;
  const u16* ntcol = ntbf + (c0 + (lane & 15)) * 32;
  v16bf bT = load_b_frag(ntcol, lane, 0);
  v16bf bS = load_b_frag(nscol, lane, 0);
  v8f z = {};
  v8f p1 = __builtin_amdgcn_wmma_f32_16x16x32_bf16(false, aS, false, bT, (short)0, z, false, false);
  v8f p2 = __builtin_amdgcn_wmma_f32_16x16x32_bf16(false, aT, false, bS, (short)0, z, false, false);
  int rh = (lane >> 4) & 1;
#pragma unroll
  for (int r = 0; r < 8; ++r) {
    float a = p1[r] - p2[r];
    float v = (a > 0.0f) ? tanhf(3.0f * a) : 0.0f;   // relu(tanh(3a))
    adj[(size_t)(r0 + r + rh * 8) * N + c0 + (lane & 15)] = v;
  }
}

// ---------------------------------------------------------------------------
// 3) Top-30 per row of (adj + 0.01*noise): 30x argmax tree in LDS, mask in place.
// ---------------------------------------------------------------------------
__global__ void topk_mask_kernel(float* __restrict__ adj, const float* __restrict__ noise) {
  const int N = N_NODES;
  __shared__ float vals[N_NODES];
  __shared__ float av[N_NODES];
  __shared__ int   flags[N_NODES];
  __shared__ float smax[256];
  __shared__ int   sidx[256];
  int row = blockIdx.x, tid = threadIdx.x;
  for (int j = tid; j < N; j += 256) {
    float a = adj[(size_t)row * N + j];
    av[j] = a;
    vals[j] = a + 0.01f * noise[(size_t)row * N + j];
    flags[j] = 0;
  }
  __syncthreads();
  for (int it = 0; it < 30; ++it) {
    float best = -3.4e38f; int bi = 0;
    for (int j = tid; j < N; j += 256) { float v = vals[j]; if (v > best) { best = v; bi = j; } }
    smax[tid] = best; sidx[tid] = bi;
    __syncthreads();
    for (int s = 128; s > 0; s >>= 1) {
      if (tid < s && smax[tid + s] > smax[tid]) { smax[tid] = smax[tid + s]; sidx[tid] = sidx[tid + s]; }
      __syncthreads();
    }
    if (tid == 0) { flags[sidx[0]] = 1; vals[sidx[0]] = -3.4e38f; }
    __syncthreads();
  }
  for (int j = tid; j < N; j += 256)
    adj[(size_t)row * N + j] = flags[j] ? av[j] : 0.0f;
}

// ---------------------------------------------------------------------------
// 4) adjT = transpose(adj), LDS-tiled.
// ---------------------------------------------------------------------------
__global__ void transpose_kernel(const float* __restrict__ in, float* __restrict__ out) {
  const int N = N_NODES;
  __shared__ float tile[32][33];
  int bx = blockIdx.x * 32, by = blockIdx.y * 32;
  int tx = threadIdx.x, ty = threadIdx.y;   // (32,8)
#pragma unroll
  for (int i = 0; i < 32; i += 8) tile[ty + i][tx] = in[(size_t)(by + ty + i) * N + bx + tx];
  __syncthreads();
#pragma unroll
  for (int i = 0; i < 32; i += 8) out[(size_t)(bx + ty + i) * N + by + tx] = tile[tx][ty + i];
}

// ---------------------------------------------------------------------------
// 5) BN stats zero + fc1 (relu) with per-wave reduction + atomics.
// ---------------------------------------------------------------------------
__global__ void zero_stats_kernel(float* __restrict__ stats) { stats[threadIdx.x] = 0.0f; }

__global__ void fc1_stats_kernel(const float* __restrict__ hist, const float* __restrict__ w1,
                                 const float* __restrict__ b1, float* __restrict__ h,
                                 float* __restrict__ stats) {
  int tid = threadIdx.x;
  int r = blockIdx.x * 32 + (tid & 31);   // row in [0,16384)
  int c = blockIdx.y * 8  + (tid >> 5);   // channel in [0,128)
  int bb = r >> 11, n = r & (N_NODES - 1);
  float acc = b1[c];
#pragma unroll
  for (int s = 0; s < SEQ; ++s) acc += hist[(size_t)(bb * SEQ + s) * N_NODES + n] * w1[c * SEQ + s];
  float v = acc > 0.0f ? acc : 0.0f;
  h[(size_t)r * C2H + c] = v;
  float sum = v, sq = v * v;                 // one wave == one channel group
#pragma unroll
  for (int o = 16; o > 0; o >>= 1) { sum += __shfl_down(sum, o, 32); sq += __shfl_down(sq, o, 32); }
  if ((tid & 31) == 0) { atomicAdd(&stats[c], sum); atomicAdd(&stats[C2H + c], sq); }
}

// ---------------------------------------------------------------------------
// 6) BN apply + fc2 -> dy_feat (16384 x 64). One row per 128-thread block.
// ---------------------------------------------------------------------------
__global__ void bn_fc2_kernel(const float* __restrict__ h, const float* __restrict__ stats,
                              const float* __restrict__ gamma, const float* __restrict__ beta,
                              const float* __restrict__ w2, const float* __restrict__ b2,
                              float* __restrict__ dy) {
  __shared__ float hn[C2H];
  int r = blockIdx.x, tid = threadIdx.x;    // 128 threads
  float x  = h[(size_t)r * C2H + tid];
  float mu = stats[tid] * (1.0f / (float)ROWS);
  float var = stats[C2H + tid] * (1.0f / (float)ROWS) - mu * mu;
  hn[tid] = (x - mu) * rsqrtf(var + 1e-5f) * gamma[tid] + beta[tid];
  __syncthreads();
  if (tid < HID) {
    float acc = b2[tid];
    const float* wr = w2 + tid * C2H;
#pragma unroll 4
    for (int c = 0; c < C2H; ++c) acc += hn[c] * wr[c];
    dy[(size_t)r * HID + tid] = acc;
  }
}

// ---------------------------------------------------------------------------
// 7) Q/K projections for both maps (share the 128 common feature channels).
//    feat = [dy(64) | tid(32) | diw(32) | ns-or-nt(32)], wq/wk are (64 x 160).
// ---------------------------------------------------------------------------
__global__ void qk_proj_kernel(const float* __restrict__ dy, const float* __restrict__ tidf,
                               const float* __restrict__ diwf, const float* __restrict__ nsf,
                               const float* __restrict__ ntf, const float* __restrict__ wq,
                               const float* __restrict__ wk,
                               u16* __restrict__ Q1, u16* __restrict__ K1,
                               u16* __restrict__ Q2, u16* __restrict__ K2) {
  __shared__ float fc[C2H];
  __shared__ float sn[32];
  __shared__ float st[32];
  int r = blockIdx.x, t = threadIdx.x;      // 64 threads, one (b,n) row
  int b = r >> 11, n = r & (N_NODES - 1);
  size_t lastT = ((size_t)(b * SEQ + (SEQ - 1)) * N_NODES + n) * 32;
  for (int i = t; i < C2H; i += 64) {
    float v;
    if (i < 64)      v = dy[(size_t)r * HID + i];
    else if (i < 96) v = tidf[lastT + (i - 64)];
    else             v = diwf[lastT + (i - 96)];
    fc[i] = v;
  }
  if (t < 32) { sn[t] = nsf[n * 32 + t]; st[t] = ntf[n * 32 + t]; }
  __syncthreads();
  const float* wqr = wq + t * 160;
  const float* wkr = wk + t * 160;
  float qc = 0.0f, kc = 0.0f;
#pragma unroll 4
  for (int c = 0; c < C2H; ++c) { float f = fc[c]; qc += f * wqr[c]; kc += f * wkr[c]; }
  float qs = 0.0f, qt = 0.0f, ks = 0.0f, kt = 0.0f;
#pragma unroll
  for (int c = 0; c < 32; ++c) {
    float a = sn[c], bb = st[c], wqc = wqr[128 + c], wkc = wkr[128 + c];
    qs += a * wqc; qt += bb * wqc; ks += a * wkc; kt += bb * wkc;
  }
  Q1[(size_t)r * HID + t] = f2bf(qc + qs);
  Q2[(size_t)r * HID + t] = f2bf(qc + qt);
  K1[(size_t)r * HID + t] = f2bf(kc + ks);
  K2[(size_t)r * HID + t] = f2bf(kc + kt);
}

// ---------------------------------------------------------------------------
// 8) W = softmax(Q K^T / 8): the bandwidth-critical kernel.
//    Block = 8 waves; owns 16 rows x all 2048 cols. Full 16x2048 f32 logits
//    tile held in LDS (128 KB dynamic shared; CDNA5 WGP has 320 KB) so the
//    output is produced in a single streaming-write pass. Each wave computes
//    a 16x16 tile per step via two v_wmma_f32_16x16x32_bf16 (K=64). K-matrix
//    (256 KB per (b,map)) streams through L2 (192 MB) shared by all 128
//    row-block workgroups -> no LDS staging needed for K.
// ---------------------------------------------------------------------------
__global__ void attn_kernel(const u16* __restrict__ Q1, const u16* __restrict__ K1,
                            const u16* __restrict__ Q2, const u16* __restrict__ K2,
                            float* __restrict__ out /* base of W1 */) {
  const int N = N_NODES;
  extern __shared__ float logits[];          // 16 * 2048 floats = 128 KB
  __shared__ float s_red[256];
  __shared__ float s_rmax[16];
  __shared__ float s_rsum[16];
  int tid = threadIdx.x, lane = tid & 31, w = tid >> 5;
  int r0 = blockIdx.x * 16, b = blockIdx.y, z = blockIdx.z;
  const u16* Q = z ? Q2 : Q1;
  const u16* K = z ? K2 : K1;

  const u16* qrow = Q + (size_t)(b * N + r0 + (lane & 15)) * HID;
  v16bf A0 = load_a_frag(qrow, lane, 0);     // K = 0..31
  v16bf A1 = load_a_frag(qrow, lane, 32);    // K = 32..63
  const float scale = 0.125f;                // 1/sqrt(64)
  int rh = (lane >> 4) & 1;

  for (int it = 0; it < N / 128; ++it) {
    int c0 = it * 128 + w * 16;
    const u16* krow = K + (size_t)(b * N + c0 + (lane & 15)) * HID;
    v16bf B0 = load_b_frag(krow, lane, 0);
    v16bf B1 = load_b_frag(krow, lane, 32);
    v8f acc = {};
    acc = __builtin_amdgcn_wmma_f32_16x16x32_bf16(false, A0, false, B0, (short)0, acc, false, false);
    acc = __builtin_amdgcn_wmma_f32_16x16x32_bf16(false, A1, false, B1, (short)0, acc, false, false);
#pragma unroll
    for (int r = 0; r < 8; ++r)
      logits[(r + rh * 8) * N + c0 + (lane & 15)] = acc[r] * scale;
  }
  __syncthreads();

  // Softmax over each of the 16 rows (16 threads per row).
  int row = tid >> 4, lir = tid & 15;
  float m = -3.4e38f;
  for (int j = lir; j < N; j += 16) { float v = logits[row * N + j]; m = v > m ? v : m; }
  s_red[tid] = m;
  __syncthreads();
  if (lir == 0) {
    float mm = s_red[tid];
#pragma unroll
    for (int k = 1; k < 16; ++k) mm = fmaxf(mm, s_red[tid + k]);
    s_rmax[row] = mm;
  }
  __syncthreads();
  float rm = s_rmax[row], sm = 0.0f;
  for (int j = lir; j < N; j += 16) {
    float e = __expf(logits[row * N + j] - rm);
    logits[row * N + j] = e;
    sm += e;
  }
  s_red[tid] = sm;
  __syncthreads();
  if (lir == 0) {
    float ss = 0.0f;
#pragma unroll
    for (int k = 0; k < 16; ++k) ss += s_red[tid + k];
    s_rsum[row] = 1.0f / ss;
  }
  __syncthreads();

  // Coalesced streaming write of the normalized tile.
  float* ob = out + ((size_t)z * BATCH + b) * (size_t)N * N + (size_t)r0 * N;
  for (int idx = tid; idx < 16 * N; idx += 256) {
    int rr = idx >> 11, cc = idx & (N - 1);
    ob[(size_t)rr * N + cc] = logits[idx] * s_rsum[rr];
  }
}

// ---------------------------------------------------------------------------
// Launch
// ---------------------------------------------------------------------------
extern "C" void kernel_launch(void* const* d_in, const int* in_sizes, int n_in,
                              void* d_out, int out_size, void* d_ws, size_t ws_size,
                              hipStream_t stream) {
  (void)in_sizes; (void)n_in; (void)out_size; (void)ws_size;
  const float* hist  = (const float*)d_in[0];
  const float* embs  = (const float*)d_in[1];
  const float* embt  = (const float*)d_in[2];
  const float* tidf  = (const float*)d_in[3];
  const float* diwf  = (const float*)d_in[4];
  const float* noise = (const float*)d_in[5];
  const float* l1w   = (const float*)d_in[6];
  const float* l1b   = (const float*)d_in[7];
  const float* l2w   = (const float*)d_in[8];
  const float* l2b   = (const float*)d_in[9];
  const float* f1w   = (const float*)d_in[10];
  const float* f1b   = (const float*)d_in[11];
  const float* f2w   = (const float*)d_in[12];
  const float* f2b   = (const float*)d_in[13];
  const float* gam   = (const float*)d_in[14];
  const float* bet   = (const float*)d_in[15];
  const float* wq    = (const float*)d_in[16];
  const float* wk    = (const float*)d_in[17];

  const size_t NN = (size_t)N_NODES * N_NODES;
  float* out_adj  = (float*)d_out;
  float* out_adjT = out_adj + NN;
  float* out_W    = out_adjT + NN;    // W1 then W2, each BATCH*NN

  // Workspace layout (f32 region then 32B-aligned bf16 region).
  float* nsf   = (float*)d_ws;                     // 2048*32
  float* ntf   = nsf + N_NODES * 32;               // 2048*32
  float* h     = ntf + N_NODES * 32;               // 16384*128
  float* dy    = h + (size_t)ROWS * C2H;           // 16384*64
  float* stats = dy + (size_t)ROWS * HID;          // 256
  u16* nsbf = (u16*)(stats + 256);                 // 32B-aligned by construction
  u16* ntbf = nsbf + N_NODES * 32;
  u16* Q1   = ntbf + N_NODES * 32;                 // each 16384*64
  u16* K1   = Q1 + (size_t)ROWS * HID;
  u16* Q2   = K1 + (size_t)ROWS * HID;
  u16* K2   = Q2 + (size_t)ROWS * HID;

  // 1) node embeddings
  embed_kernel<<<N_NODES, 64, 0, stream>>>(embs, embt, l1w, l1b, l2w, l2b,
                                           nsf, ntf, nsbf, ntbf);
  // 2) adjacency GEMM (WMMA bf16)
  adj_gemm_kernel<<<dim3(N_NODES / 128, N_NODES / 16), 256, 0, stream>>>(nsbf, ntbf, out_adj);
  // 3) top-30 mask in place
  topk_mask_kernel<<<N_NODES, 256, 0, stream>>>(out_adj, noise);
  // 4) transpose
  transpose_kernel<<<dim3(N_NODES / 32, N_NODES / 32), dim3(32, 8), 0, stream>>>(out_adj, out_adjT);
  // 5) fc1 + BN stats
  zero_stats_kernel<<<1, 256, 0, stream>>>(stats);
  fc1_stats_kernel<<<dim3(ROWS / 32, C2H / 8), 256, 0, stream>>>(hist, f1w, f1b, h, stats);
  // 6) BN + fc2
  bn_fc2_kernel<<<ROWS, C2H, 0, stream>>>(h, stats, gam, bet, f2w, f2b, dy);
  // 7) Q/K projections (bf16)
  qk_proj_kernel<<<ROWS, 64, 0, stream>>>(dy, tidf, diwf, nsf, ntf, wq, wk, Q1, K1, Q2, K2);
  // 8) attention maps W1/W2 (WMMA bf16 + LDS-resident softmax)
  const int attn_lds = 16 * N_NODES * (int)sizeof(float);   // 128 KB (CDNA5 WGP: 320 KB)
  (void)hipFuncSetAttribute((const void*)attn_kernel,
                            hipFuncAttributeMaxDynamicSharedMemorySize, attn_lds);
  attn_kernel<<<dim3(N_NODES / 16, BATCH, 2), 256, attn_lds, stream>>>(Q1, K1, Q2, K2, out_W);
}
